// TemporalFusionModule_60730837565462
// MI455X (gfx1250) — compile-verified, matrix operand order checked by reading, not run on hardware
//
#include <hip/hip_runtime.h>
#include <math.h>

typedef __attribute__((ext_vector_type(16))) _Float16 v16h;
typedef __attribute__((ext_vector_type(8)))  _Float16 v8h;
typedef __attribute__((ext_vector_type(2)))  _Float16 v2h;
typedef __attribute__((ext_vector_type(8)))  float    v8f;

#define BATCH   8
#define TLEN    32
#define NTOK    (BATCH * TLEN)       // 256 tokens
#define DMODEL  256
#define DINNER  512
#define DSTATE  16
#define DTRANK  16

#define ACT_NONE     0
#define ACT_RELU     1
#define ACT_SOFTPLUS 2
#define ACT_SIGMOID  3

// -------------------------------------------------------------------------
// Unconditional global fragment load (f32 -> f16) for the B operand.
// ISA 16-bit fragment layout: lanes 0-15 hold K {0..7, 16..23} of the chunk,
// lanes 16-31 hold K {8..15, 24..31}; caller passes p = row + k0 + 8*half.
// Two 8-contiguous-float runs -> 4x global_load_b128 + v_cvt_pk_f16_f32.
// -------------------------------------------------------------------------
__device__ __forceinline__ v16h load_frag_f32(const float* __restrict__ p) {
  const float4 a0 = *(const float4*)(p + 0);
  const float4 a1 = *(const float4*)(p + 4);
  const float4 b0 = *(const float4*)(p + 16);
  const float4 b1 = *(const float4*)(p + 20);
  v16h v;
  v[0]  = (_Float16)a0.x; v[1]  = (_Float16)a0.y;
  v[2]  = (_Float16)a0.z; v[3]  = (_Float16)a0.w;
  v[4]  = (_Float16)a1.x; v[5]  = (_Float16)a1.y;
  v[6]  = (_Float16)a1.z; v[7]  = (_Float16)a1.w;
  v[8]  = (_Float16)b0.x; v[9]  = (_Float16)b0.y;
  v[10] = (_Float16)b0.z; v[11] = (_Float16)b0.w;
  v[12] = (_Float16)b1.x; v[13] = (_Float16)b1.y;
  v[14] = (_Float16)b1.z; v[15] = (_Float16)b1.w;
  return v;
}

// A fragment from an LDS-staged 16x32 f16 chunk: two ds_load_b128 per lane.
__device__ __forceinline__ v16h lds_frag(const _Float16* __restrict__ ap) {
  v8h lo = *(const v8h*)(ap);        // K run {8*half + 0..7}
  v8h hi = *(const v8h*)(ap + 16);   // K run {16 + 8*half + 0..7}
  v16h v;
#pragma unroll
  for (int i = 0; i < 8; ++i) { v[i] = lo[i]; v[8 + i] = hi[i]; }
  return v;
}

// -------------------------------------------------------------------------
// LDS-staged WMMA GEMM:  C(M,N) = act( A(M,K) @ W(N,K)^T + bias + resid )
// Block = 256 threads = 8 waves = one 16(M) x 128(N) macro-tile.
// The 16x32 A chunk is staged once per block into LDS (f16, double buffered,
// one barrier per K-chunk); every wave broadcasts-reads the same fragment.
// Preconditions: M % 16 == 0 (exact grid), K % 32 == 0, lda/ldw % 4 == 0.
// -------------------------------------------------------------------------
__global__ __launch_bounds__(256)
void wmma_gemm_kernel(const float* __restrict__ A, int lda,
                      const float* __restrict__ W, int ldw,
                      const float* __restrict__ bias,
                      const float* __restrict__ resid, int ldr,
                      float* __restrict__ C, int ldc,
                      int M, int N, int K, int act) {
  __shared__ _Float16 As[2][512];
  const int tid  = threadIdx.x;
  const int lane = tid & 31;
  const int wave = tid >> 5;
  const int nt_n  = (N + 15) >> 4;
  const int ntb_n = (nt_n + 7) >> 3;
  const int tm = (blockIdx.x / ntb_n) << 4;
  const int tn = ((((blockIdx.x % ntb_n) << 3) + wave) << 4);
  const int half = lane >> 4;
  const int l15  = lane & 15;
  const int wn = tn + l15;
  const float* wrow = W + (size_t)(wn < N ? wn : 0) * ldw + 8 * half;
  // staging map: thread -> 2 consecutive elements of the 16x32 chunk
  const int se = tid << 1;
  const int sr = se >> 5;                    // row within the M-tile
  const int sk = se & 31;                    // k within the chunk
  const float* srow = A + (size_t)(tm + sr) * lda + sk;   // always in range
  const _Float16* ap = &As[0][l15 * 32 + 8 * half];

  v8f acc = {};
  int bufi = 0;
  for (int k0 = 0; k0 < K; k0 += 32, bufi ^= 1) {
    float2 g = *(const float2*)(srow + k0);
    v2h pk; pk[0] = (_Float16)g.x; pk[1] = (_Float16)g.y;
    *(v2h*)(&As[bufi][se]) = pk;
    __syncthreads();
    v16h af = lds_frag(ap + bufi * 512);
    v16h bf = load_frag_f32(wrow + k0);
    acc = __builtin_amdgcn_wmma_f32_16x16x32_f16(false, af, false, bf,
                                                 (short)0, acc, false, false);
  }
  // epilogue: branch-free loads (uniform null checks, clamped column), guarded store
  const int n   = tn + l15;
  const int ncl = (n < N) ? n : 0;
  const bool nok = (n < N);
  const float bv = bias ? bias[ncl] : 0.0f;
#pragma unroll
  for (int r = 0; r < 8; ++r) {
    int m = tm + r + 8 * half;               // always < M (M % 16 == 0)
    float v = acc[r] + bv;
    if (resid) v += resid[(size_t)m * ldr + ncl];
    if (act == ACT_RELU)          v = fmaxf(v, 0.0f);
    else if (act == ACT_SOFTPLUS) v = (v > 20.0f) ? v : log1pf(__expf(v));
    else if (act == ACT_SIGMOID)  v = 1.0f / (1.0f + __expf(-v));
    if (nok) C[(size_t)m * ldc + n] = v;
  }
}

// -------------------------------------------------------------------------
// Implicit-GEMM 3D conv, stride (1,2,2), pad 1, kernel 3x3x3, fused
// bias + BN affine + ReLU.  Block = one (b,t,mt) tile; waves = Cout/16
// N-tiles (blockDim = Cout*2).  The branch-free im2col gather for the 16x32
// A chunk is done ONCE per block into LDS (amortized over all Cout tiles),
// then each wave broadcast-reads its fragment (ds_load_b128) and loads its
// weight fragment from global (rows padded to KKpad, stride KKpad).
// frames_layout: 1 -> input is (B,T,C,H,W), 0 -> (B,C,T,H,W).
// Output written NCDHW: (B,Cout,T,OH,OW).
// -------------------------------------------------------------------------
__global__ __launch_bounds__(256)
void conv3d_wmma_kernel(const float* __restrict__ in,
                        const float* __restrict__ w,   // (Cout, KKpad)
                        const float* __restrict__ cb,
                        const float* __restrict__ bn_g,
                        const float* __restrict__ bn_b,
                        float* __restrict__ out,
                        int Cin, int Cout, int IH, int IW, int OH, int OW,
                        int frames_layout) {
  __shared__ _Float16 As[2][512];
  const int tid  = threadIdx.x;
  const int lane = tid & 31;
  const int wave = tid >> 5;                 // cout tile index
  const int KK    = Cin * 27;
  const int KKpad = (KK + 31) & ~31;
  const int OP = OH * OW;                    // multiple of 16 for all layers
  const int nt_m = OP >> 4;
  const int mt    = blockIdx.x % nt_m;       // grid exact: planes * nt_m
  const int plane = blockIdx.x / nt_m;
  const int b = plane / TLEN;
  const int t = plane % TLEN;
  const int half = lane >> 4;
  const int l15  = lane & 15;
  const int co = (wave << 4) + l15;          // always < Cout
  const float* wrow = w + (size_t)co * KKpad + 8 * half;
  const _Float16* ap = &As[0][l15 * 32 + 8 * half];

  v8f acc = {};
  int bufi = 0;
  for (int k0 = 0; k0 < KKpad; k0 += 32, bufi ^= 1) {
    for (int e = tid * 2; e < 512; e += (int)blockDim.x * 2) {
      int r  = e >> 5;                       // row within M-tile
      int p  = (mt << 4) + r;
      int oh = p / OW, ow = p % OW;
      v2h pk;
#pragma unroll
      for (int u = 0; u < 2; ++u) {          // branch-free im2col gather
        int k = k0 + (e & 31) + u;
        int ci = k / 27, rr = k % 27;
        int kd = rr / 9, kh = (rr % 9) / 3, kw = rr % 3;
        int id = t + kd - 1;
        int ih = oh * 2 + kh - 1;
        int iw = ow * 2 + kw - 1;
        bool ok = (k < KK) & ((unsigned)id < (unsigned)TLEN) &
                  ((unsigned)ih < (unsigned)IH) & ((unsigned)iw < (unsigned)IW);
        int cic = ok ? ci : 0;               // clamped, always-valid coords
        int idc = ok ? id : 0;
        int ihc = ok ? ih : 0;
        int iwc = ok ? iw : 0;
        size_t off = frames_layout
          ? ((((size_t)(b * TLEN + idc) * Cin + cic) * IH + ihc) * IW + iwc)
          : ((((size_t)(b * Cin + cic) * TLEN + idc) * IH + ihc) * IW + iwc);
        float f = in[off];                   // unconditional load
        pk[u] = (_Float16)(ok ? f : 0.0f);   // cndmask
      }
      *(v2h*)(&As[bufi][e]) = pk;            // ds_store_b32
    }
    __syncthreads();
    v16h af = lds_frag(ap + bufi * 512);
    v16h bf = load_frag_f32(wrow + k0);
    acc = __builtin_amdgcn_wmma_f32_16x16x32_f16(false, af, false, bf,
                                                 (short)0, acc, false, false);
  }
  const float bc = cb[co], gc = bn_g[co], bec = bn_b[co];
#pragma unroll
  for (int r = 0; r < 8; ++r) {
    int m = (mt << 4) + r + 8 * half;        // spatial pos (always < OP)
    float v = gc * (acc[r] + bc) + bec;
    v = fmaxf(v, 0.0f);
    out[(((size_t)(b * Cout + co) * TLEN + t) * OP) + m] = v;
  }
}

// ------------------------- small helper kernels --------------------------
// Zero-padded row copy: dst(rows, dstCols) <- src(rows, srcStride)[:, :srcCols]
__global__ void pad_rows_kernel(const float* __restrict__ src, int srcStride,
                                int srcCols, float* __restrict__ dst,
                                int dstCols, int rows) {
  int idx = blockIdx.x * blockDim.x + threadIdx.x;
  if (idx >= rows * dstCols) return;
  int c = idx % dstCols, r = idx / dstCols;
  dst[idx] = (c < srcCols) ? src[(size_t)r * srcStride + c] : 0.0f;
}

__global__ void maxpool2x2_kernel(const float* __restrict__ in,
                                  float* __restrict__ out, int C, int H, int W) {
  const int OW = W >> 1, OH = H >> 1;
  const size_t total = (size_t)BATCH * C * TLEN * OH * OW;
  size_t idx = (size_t)blockIdx.x * blockDim.x + threadIdx.x;
  if (idx >= total) return;
  int ow = (int)(idx % OW);
  int oh = (int)((idx / OW) % OH);
  size_t rest = idx / ((size_t)OH * OW);     // (b*C + c)*T + t
  const float* p = in + (rest * H + (size_t)oh * 2) * W + (size_t)ow * 2;
  out[idx] = fmaxf(fmaxf(p[0], p[1]), fmaxf(p[W], p[W + 1]));
}

// (B,C,T,2,2) mean over spatial -> token-major (b*T+t, C)
__global__ void spatial_mean_kernel(const float* __restrict__ in,
                                    float* __restrict__ out, int C) {
  int idx = blockIdx.x * blockDim.x + threadIdx.x;
  if (idx >= NTOK * C) return;
  int c = idx % C, tok = idx / C;
  int b = tok / TLEN, t = tok % TLEN;
  const float* p = in + ((size_t)(b * C + c) * TLEN + t) * 4;
  out[idx] = 0.25f * (p[0] + p[1] + p[2] + p[3]);
}

// depthwise causal conv1d (kernel 4, left pad 3) + SiLU; xz token-major 1024 wide
__global__ void conv1d_silu_kernel(const float* __restrict__ xz,
                                   const float* __restrict__ w,
                                   const float* __restrict__ cb,
                                   float* __restrict__ xs) {
  int idx = blockIdx.x * blockDim.x + threadIdx.x;
  if (idx >= NTOK * DINNER) return;
  int d = idx % DINNER, tok = idx / DINNER;
  int b = tok / TLEN, t = tok % TLEN;
  float s = cb[d];
#pragma unroll
  for (int j = 0; j < 4; ++j) {
    int ts = t - 3 + j;
    int tc = ts >= 0 ? ts : 0;
    float xv = xz[(size_t)(b * TLEN + tc) * (2 * DINNER) + d];  // unconditional
    s += (ts >= 0 ? xv : 0.0f) * w[d * 4 + j];
  }
  xs[idx] = s / (1.0f + __expf(-s));          // silu
}

// selective scan: one thread per (b, d) row; T=32 sequential, 16 states
__global__ void ssm_scan_kernel(const float* __restrict__ dtb,
                                const float* __restrict__ xs,
                                const float* __restrict__ xdbl,   // tok x 48
                                const float* __restrict__ xz,
                                const float* __restrict__ A_log,
                                const float* __restrict__ Dp,
                                float* __restrict__ yb) {
  int idx = blockIdx.x * blockDim.x + threadIdx.x;
  if (idx >= BATCH * DINNER) return;
  int d = idx % DINNER, b = idx / DINNER;
  float Ar[DSTATE], h[DSTATE];
#pragma unroll
  for (int s = 0; s < DSTATE; ++s) { Ar[s] = -__expf(A_log[d * DSTATE + s]); h[s] = 0.0f; }
  const float Dd = Dp[d];
  for (int t = 0; t < TLEN; ++t) {
    int tok = b * TLEN + t;
    float dt = dtb[(size_t)tok * DINNER + d];
    float xv = xs[(size_t)tok * DINNER + d];
    const float* xrow = xdbl + (size_t)tok * 48;
    float y = Dd * xv;
#pragma unroll
    for (int s = 0; s < DSTATE; ++s) {
      h[s] = __expf(dt * Ar[s]) * h[s] + dt * xrow[16 + s] * xv;
      y += h[s] * xrow[32 + s];
    }
    float zv = xz[(size_t)tok * (2 * DINNER) + DINNER + d];
    y *= zv / (1.0f + __expf(-zv));
    yb[(size_t)tok * DINNER + d] = y;
  }
}

// Kalman filter: one thread per batch, T=32 sequential; 4x4 solve via Gauss.
__global__ void kalman_kernel(const float* __restrict__ tb,
                              const float* __restrict__ meas,
                              const float* __restrict__ F,
                              const float* __restrict__ Hm,
                              const float* __restrict__ logq,
                              const float* __restrict__ logr,
                              float* __restrict__ pred) {
  int b = blockIdx.x * blockDim.x + threadIdx.x;
  if (b >= BATCH) return;
  float Qd[4], Rd[4], s[4], P[16];
  for (int i = 0; i < 4; ++i) { Qd[i] = __expf(logq[i]) + 1e-6f; Rd[i] = __expf(logr[i]) + 1e-6f; }
  for (int i = 0; i < 4; ++i) s[i] = tb[(size_t)(b * TLEN) * 4 + i];
  for (int i = 0; i < 16; ++i) P[i] = 0.0f;
  for (int i = 0; i < 4; ++i) P[i * 4 + i] = 0.01f;

  for (int t = 0; t < TLEN; ++t) {
    float sp[4], FP[16], Pp[16], yr[4], HP[16], S[16];
    for (int i = 0; i < 4; ++i) {
      float a = 0.0f;
      for (int j = 0; j < 4; ++j) a += F[i * 4 + j] * s[j];
      sp[i] = a;
    }
    for (int i = 0; i < 4; ++i)
      for (int j = 0; j < 4; ++j) {
        float a = 0.0f;
        for (int k = 0; k < 4; ++k) a += F[i * 4 + k] * P[k * 4 + j];
        FP[i * 4 + j] = a;
      }
    for (int i = 0; i < 4; ++i)
      for (int j = 0; j < 4; ++j) {
        float a = 0.0f;
        for (int k = 0; k < 4; ++k) a += FP[i * 4 + k] * F[j * 4 + k];
        Pp[i * 4 + j] = a + ((i == j) ? Qd[i] : 0.0f);
      }
    for (int i = 0; i < 4; ++i) {
      float a = 0.0f;
      for (int j = 0; j < 4; ++j) a += Hm[i * 4 + j] * sp[j];
      yr[i] = meas[(size_t)(b * TLEN + t) * 4 + i] - a;
    }
    for (int i = 0; i < 4; ++i)
      for (int j = 0; j < 4; ++j) {
        float a = 0.0f;
        for (int k = 0; k < 4; ++k) a += Hm[i * 4 + k] * Pp[k * 4 + j];
        HP[i * 4 + j] = a;
      }
    for (int i = 0; i < 4; ++i)
      for (int j = 0; j < 4; ++j) {
        float a = 0.0f;
        for (int k = 0; k < 4; ++k) a += HP[i * 4 + k] * Hm[j * 4 + k];
        S[i * 4 + j] = a + ((i == j) ? Rd[i] : 0.0f);
      }
    float Aug[4][8];
    for (int i = 0; i < 4; ++i)
      for (int j = 0; j < 4; ++j) { Aug[i][j] = S[i * 4 + j]; Aug[i][4 + j] = Hm[i * 4 + j]; }
    for (int col = 0; col < 4; ++col) {
      int piv = col; float best = fabsf(Aug[col][col]);
      for (int r = col + 1; r < 4; ++r) {
        float v = fabsf(Aug[r][col]);
        if (v > best) { best = v; piv = r; }
      }
      if (piv != col)
        for (int j = 0; j < 8; ++j) { float tmp = Aug[col][j]; Aug[col][j] = Aug[piv][j]; Aug[piv][j] = tmp; }
      float inv = 1.0f / Aug[col][col];
      for (int j = 0; j < 8; ++j) Aug[col][j] *= inv;
      for (int r = 0; r < 4; ++r) if (r != col) {
        float f2 = Aug[r][col];
        for (int j = 0; j < 8; ++j) Aug[r][j] -= f2 * Aug[col][j];
      }
    }
    float Kg[16], IKH[16], T1[16], Pn[16];
    for (int i = 0; i < 4; ++i)                 // K = Pp @ X^T
      for (int j = 0; j < 4; ++j) {
        float a = 0.0f;
        for (int k = 0; k < 4; ++k) a += Pp[i * 4 + k] * Aug[j][4 + k];
        Kg[i * 4 + j] = a;
      }
    for (int i = 0; i < 4; ++i) {               // s_new = sp + K yr
      float a = sp[i];
      for (int j = 0; j < 4; ++j) a += Kg[i * 4 + j] * yr[j];
      s[i] = a;
    }
    for (int i = 0; i < 4; ++i)                 // IKH = I - K H
      for (int j = 0; j < 4; ++j) {
        float a = (i == j) ? 1.0f : 0.0f;
        for (int k = 0; k < 4; ++k) a -= Kg[i * 4 + k] * Hm[k * 4 + j];
        IKH[i * 4 + j] = a;
      }
    for (int i = 0; i < 4; ++i)
      for (int j = 0; j < 4; ++j) {
        float a = 0.0f;
        for (int k = 0; k < 4; ++k) a += IKH[i * 4 + k] * Pp[k * 4 + j];
        T1[i * 4 + j] = a;
      }
    for (int i = 0; i < 4; ++i)
      for (int j = 0; j < 4; ++j) {
        float a = 0.0f;
        for (int k = 0; k < 4; ++k) a += T1[i * 4 + k] * IKH[j * 4 + k] + Kg[i * 4 + k] * Rd[k] * Kg[j * 4 + k];
        Pn[i * 4 + j] = a;
      }
    for (int i = 0; i < 16; ++i) P[i] = Pn[i];
    for (int i = 0; i < 4; ++i) pred[(size_t)(b * TLEN + t) * 4 + i] = s[i];
  }
}

// -------------------------------------------------------------------------
extern "C" void kernel_launch(void* const* d_in, const int* in_sizes, int n_in,
                              void* d_out, int out_size, void* d_ws, size_t ws_size,
                              hipStream_t stream) {
  (void)in_sizes; (void)n_in; (void)out_size;
  const float* frames  = (const float*)d_in[0];
  const float* tb      = (const float*)d_in[1];
  const float* cw0 = (const float*)d_in[2];  const float* cb0 = (const float*)d_in[3];
  const float* bg0 = (const float*)d_in[4];  const float* bb0 = (const float*)d_in[5];
  const float* cw1 = (const float*)d_in[6];  const float* cb1 = (const float*)d_in[7];
  const float* bg1 = (const float*)d_in[8];  const float* bb1 = (const float*)d_in[9];
  const float* cw2 = (const float*)d_in[10]; const float* cb2 = (const float*)d_in[11];
  const float* bg2 = (const float*)d_in[12]; const float* bb2 = (const float*)d_in[13];
  const float* ep_w = (const float*)d_in[14]; const float* ep_b = (const float*)d_in[15];
  const float* bp1_w = (const float*)d_in[16]; const float* bp1_b = (const float*)d_in[17];
  const float* bp2_w = (const float*)d_in[18]; const float* bp2_b = (const float*)d_in[19];
  const float* in_proj_w = (const float*)d_in[20];
  const float* c1d_w = (const float*)d_in[21]; const float* c1d_b = (const float*)d_in[22];
  const float* x_proj_w = (const float*)d_in[23];
  const float* dt_w = (const float*)d_in[24]; const float* dt_b = (const float*)d_in[25];
  const float* A_log = (const float*)d_in[26]; const float* Dp = (const float*)d_in[27];
  const float* out_proj_w = (const float*)d_in[28];
  const float* h1_w = (const float*)d_in[29]; const float* h1_b = (const float*)d_in[30];
  const float* h2_w = (const float*)d_in[31]; const float* h2_b = (const float*)d_in[32];
  const float* kf_F = (const float*)d_in[33]; const float* kf_H = (const float*)d_in[34];
  const float* kf_lq = (const float*)d_in[35]; const float* kf_lr = (const float*)d_in[36];

  float* outp = (float*)d_out;
  float* kf_pred = outp;             // (8,32,4) = 1024 floats
  float* meas    = outp + 1024;      // (8,32,4) = 1024 floats

  // ---------------- workspace layout (floats) ----------------
  float* ws = (float*)d_ws;
  size_t o = 0;
  float* conv0 = ws + o; o += (size_t)BATCH * 32 * TLEN * 64 * 64;   // region A (33.5M)
  float* pool0 = ws + o; o += (size_t)BATCH * 32 * TLEN * 32 * 32;   // 8.39M
  // region A reuse after pool0 consumed conv0:
  float* conv1 = conv0;                                              // 4.19M
  float* pool1 = conv1 + (size_t)BATCH * 64 * TLEN * 16 * 16;        // 1.05M
  float* conv2 = pool1 + (size_t)BATCH * 64 * TLEN * 8 * 8;          // 0.39M
  float* pool2 = conv2 + (size_t)BATCH * 96 * TLEN * 4 * 4;          // 98K
  float* u_pre = ws + o; o += (size_t)NTOK * 96;
  float* bfh   = ws + o; o += (size_t)NTOK * 64;
  float* bfv   = ws + o; o += (size_t)NTOK * DMODEL;
  float* u     = ws + o; o += (size_t)NTOK * DMODEL;
  float* xz    = ws + o; o += (size_t)NTOK * 2 * DINNER;
  float* xs    = ws + o; o += (size_t)NTOK * DINNER;
  float* xdbl  = ws + o; o += (size_t)NTOK * 48;
  float* dtb   = ws + o; o += (size_t)NTOK * DINNER;
  float* yb    = ws + o; o += (size_t)NTOK * DINNER;
  float* mo    = ws + o; o += (size_t)NTOK * DMODEL;
  float* hid   = ws + o; o += (size_t)NTOK * 128;
  // K-padded copies (so every WMMA K-loop is guard-free, K % 32 == 0):
  float* tbp   = ws + o; o += (size_t)NTOK * 32;    // tb       (256,4)  -> (256,32)
  float* bp1p  = ws + o; o += (size_t)64 * 32;      // bp1_w    (64,4)   -> (64,32)
  float* dtwp  = ws + o; o += (size_t)DINNER * 32;  // dt_w     (512,16) -> (512,32)
  float* w0p   = ws + o; o += (size_t)32 * 96;      // conv_w0  (32,81)  -> (32,96)
  float* dtin  = ws + o; o += (size_t)NTOK * 32;    // xdbl[:, :16]      -> (256,32)
  if (ws_size < o * sizeof(float)) return;

  const int TPB = 256;                         // 8 waves/block
  // GEMM grid: exact, (M/16) x ceil(ceil(N/16)/8) macro-tiles
  auto ggrid = [](int M, int N) { int ntn = (N + 15) / 16; return (M / 16) * ((ntn + 7) / 8); };
  auto eblocks = [](size_t n) { return (unsigned)((n + 255) / 256); };

  // ---- pad input-derived operands once ----
  pad_rows_kernel<<<eblocks(NTOK * 32), TPB, 0, stream>>>(tb, 4, 4, tbp, 32, NTOK);
  pad_rows_kernel<<<eblocks(64 * 32), TPB, 0, stream>>>(bp1_w, 4, 4, bp1p, 32, 64);
  pad_rows_kernel<<<eblocks(DINNER * 32), TPB, 0, stream>>>(dt_w, 16, 16, dtwp, 32, DINNER);
  pad_rows_kernel<<<eblocks(32 * 96), TPB, 0, stream>>>(cw0, 81, 81, w0p, 96, 32);

  // ---- bbox MLP (WMMA GEMMs) ----
  wmma_gemm_kernel<<<ggrid(NTOK, 64), TPB, 0, stream>>>(
      tbp, 32, bp1p, 32, bp1_b, nullptr, 0, bfh, 64, NTOK, 64, 32, ACT_RELU);
  wmma_gemm_kernel<<<ggrid(NTOK, DMODEL), TPB, 0, stream>>>(
      bfh, 64, bp2_w, 64, bp2_b, nullptr, 0, bfv, DMODEL, NTOK, DMODEL, 64, ACT_NONE);

  // ---- conv backbone: blockDim = Cout*2 (Cout/16 waves), grid = planes*nt_m ----
  {
    conv3d_wmma_kernel<<<NTOK * (64 * 64 / 16), 64, 0, stream>>>(
        frames, w0p, cb0, bg0, bb0, conv0, 3, 32, 128, 128, 64, 64, 1);
    maxpool2x2_kernel<<<eblocks((size_t)BATCH * 32 * TLEN * 32 * 32), TPB, 0, stream>>>(
        conv0, pool0, 32, 64, 64);
  }
  {
    conv3d_wmma_kernel<<<NTOK * (16 * 16 / 16), 128, 0, stream>>>(
        pool0, cw1, cb1, bg1, bb1, conv1, 32, 64, 32, 32, 16, 16, 0);
    maxpool2x2_kernel<<<eblocks((size_t)BATCH * 64 * TLEN * 8 * 8), TPB, 0, stream>>>(
        conv1, pool1, 64, 16, 16);
  }
  {
    conv3d_wmma_kernel<<<NTOK * (4 * 4 / 16), 192, 0, stream>>>(
        pool1, cw2, cb2, bg2, bb2, conv2, 64, 96, 8, 8, 4, 4, 0);
    maxpool2x2_kernel<<<eblocks((size_t)BATCH * 96 * TLEN * 2 * 2), TPB, 0, stream>>>(
        conv2, pool2, 96, 4, 4);
  }
  spatial_mean_kernel<<<eblocks(NTOK * 96), TPB, 0, stream>>>(pool2, u_pre, 96);

  // ---- embed proj + bbox residual: u = u_pre @ ep_w^T + ep_b + bf ----
  wmma_gemm_kernel<<<ggrid(NTOK, DMODEL), TPB, 0, stream>>>(
      u_pre, 96, ep_w, 96, ep_b, bfv, DMODEL, u, DMODEL, NTOK, DMODEL, 96, ACT_NONE);

  // ---- Mamba block ----
  wmma_gemm_kernel<<<ggrid(NTOK, 2 * DINNER), TPB, 0, stream>>>(
      u, DMODEL, in_proj_w, DMODEL, nullptr, nullptr, 0, xz, 2 * DINNER,
      NTOK, 2 * DINNER, DMODEL, ACT_NONE);
  conv1d_silu_kernel<<<eblocks(NTOK * DINNER), TPB, 0, stream>>>(xz, c1d_w, c1d_b, xs);
  wmma_gemm_kernel<<<ggrid(NTOK, 48), TPB, 0, stream>>>(
      xs, DINNER, x_proj_w, DINNER, nullptr, nullptr, 0, xdbl, 48,
      NTOK, 48, DINNER, ACT_NONE);
  pad_rows_kernel<<<eblocks(NTOK * 32), TPB, 0, stream>>>(xdbl, 48, 16, dtin, 32, NTOK);
  wmma_gemm_kernel<<<ggrid(NTOK, DINNER), TPB, 0, stream>>>(
      dtin, 32, dtwp, 32, dt_b, nullptr, 0, dtb, DINNER,
      NTOK, DINNER, 32, ACT_SOFTPLUS);
  ssm_scan_kernel<<<eblocks(BATCH * DINNER), TPB, 0, stream>>>(
      dtb, xs, xdbl, xz, A_log, Dp, yb);
  wmma_gemm_kernel<<<ggrid(NTOK, DMODEL), TPB, 0, stream>>>(
      yb, DINNER, out_proj_w, DINNER, nullptr, nullptr, 0, mo, DMODEL,
      NTOK, DMODEL, DINNER, ACT_NONE);

  // ---- head MLP -> meas (sigmoid), written straight into d_out[1024:] ----
  wmma_gemm_kernel<<<ggrid(NTOK, 128), TPB, 0, stream>>>(
      mo, DMODEL, h1_w, DMODEL, h1_b, nullptr, 0, hid, 128, NTOK, 128, DMODEL, ACT_RELU);
  wmma_gemm_kernel<<<ggrid(NTOK, 4), TPB, 0, stream>>>(
      hid, 128, h2_w, 128, h2_b, nullptr, 0, meas, 4, NTOK, 4, 128, ACT_SIGMOID);

  // ---- Kalman filter scan -> kf_pred in d_out[0:1024] ----
  kalman_kernel<<<1, BATCH, 0, stream>>>(tb, meas, kf_F, kf_H, kf_lq, kf_lr, kf_pred);
}
